// CLIPAttentionMKV_7327214207055
// MI455X (gfx1250) — matrix operand order repaired and checked by reading, not compiled, not run on hardware
//
#include <hip/hip_runtime.h>

// ---------------------------------------------------------------------------
// CLIP attention w/ K,V multiplier on gfx1250 (MI455X), bf16 WMMA pipeline.
// B=4 T=1024 E=1024 H=16 HD=64 MULT=2 S=2048, SCALE = 1/sqrt(64) = 0.125
// Pipeline: f32->bf16 convert; Q/K/V projections (tiled double-buffered WMMA
// GEMM); V pre-transpose to [b,h,chunk,d,s] tiles; fused flash attention with
// online softmax; output projection.
// ---------------------------------------------------------------------------

typedef __bf16 bf16;
typedef __attribute__((ext_vector_type(16))) __bf16 v16bf;
typedef __attribute__((ext_vector_type(8)))  float  v8f;

#define Bc 4
#define Tc 1024
#define Ec 1024
#define Hc 16
#define HDc 64
#define Sc 2048
#define SCALEc 0.125f

// A-matrix (16x32 bf16) element j -> k index (contiguous, matches B fills):
// lane<16: k = {0..15}; lane>=16: k = {16..31}
__device__ __forceinline__ int a_k_of(int j, int lane) {
  return (lane & 16) + j;
}

// ---------------------------------------------------------------------------
// f32 -> bf16 convert (vectorized x4)
// ---------------------------------------------------------------------------
__global__ void __launch_bounds__(256) f32_to_bf16_kernel(
    const float* __restrict__ x, bf16* __restrict__ y, int n) {
  int i = (blockIdx.x * 256 + threadIdx.x) * 4;
  if (i < n) {
    float4 v = *(const float4*)(x + i);
    y[i + 0] = (bf16)v.x;
    y[i + 1] = (bf16)v.y;
    y[i + 2] = (bf16)v.z;
    y[i + 3] = (bf16)v.w;
  }
}

// ---------------------------------------------------------------------------
// GEMM: Y[M,NN] = (X[M,1024] @ W[NN,1024]^T + bias[NN]) * scale
// Block tile 128(M) x 64(N), 8 waves, each wave 32x32 (2x2 WMMA tiles).
// Double-buffered LDS staging: one barrier per k-step; next tile's global
// loads issued before the WMMA chain of the current tile.
// ---------------------------------------------------------------------------
template <int NN, bool OUTBF>
__global__ void __launch_bounds__(256) gemm_bf16_kernel(
    const bf16* __restrict__ X, const bf16* __restrict__ W,
    const float* __restrict__ bias, void* __restrict__ outp, float scale) {
  __shared__ bf16 sX[2][128 * 32];  // 16 KB
  __shared__ bf16 sW[2][64 * 32];   //  8 KB

  const int tid = threadIdx.x;
  const int lane = tid & 31;
  const int wid  = tid >> 5;
  const int wm = wid & 3;   // 4 waves over M (32 rows each)
  const int wn = wid >> 2;  // 2 waves over N (32 cols each)
  const int bm = blockIdx.y * 128;
  const int bn = blockIdx.x * 64;
  const int mrow = lane & 15;
  const int hi = (lane >> 4) & 1;
  const int col = lane & 15;

  // staging coordinates
  const int xrow = tid >> 1, xh = tid & 1;   // X: 2 thr/row, 16 bf16 each
  const int wrow = tid >> 2, wq = tid & 3;   // W: 4 thr/row,  8 bf16 each
  const bf16* xsrc = X + (size_t)(bm + xrow) * 1024 + xh * 16;
  const bf16* wsrc = W + (size_t)(bn + wrow) * 1024 + wq * 8;

  // bias is invariant over the k loop and r rows: load once
  float bi[2];
#pragma unroll
  for (int ni = 0; ni < 2; ++ni) bi[ni] = bias[bn + wn * 32 + ni * 16 + col];

  v8f acc[2][2] = {};
  uint4 xr0, xr1, wr0;

  // preload k-step 0
  xr0 = ((const uint4*)xsrc)[0];
  xr1 = ((const uint4*)xsrc)[1];
  wr0 = *(const uint4*)wsrc;
  {
    uint4* xd = (uint4*)(sX[0] + xrow * 32 + xh * 16);
    xd[0] = xr0; xd[1] = xr1;
    *(uint4*)(sW[0] + wrow * 32 + wq * 8) = wr0;
  }
  __syncthreads();

  constexpr int KSTEPS = 1024 / 32;
  for (int it = 0; it < KSTEPS; ++it) {
    const int cur = it & 1, nxt = cur ^ 1;
    if (it + 1 < KSTEPS) {  // issue next tile's global loads early
      int kk = (it + 1) * 32;
      xr0 = ((const uint4*)(xsrc + kk))[0];
      xr1 = ((const uint4*)(xsrc + kk))[1];
      wr0 = *(const uint4*)(wsrc + kk);
    }

    v16bf a[2], b[2];
#pragma unroll
    for (int mi = 0; mi < 2; ++mi)
#pragma unroll
      for (int j = 0; j < 16; ++j)
        a[mi][j] = sX[cur][(wm * 32 + mi * 16 + mrow) * 32 + a_k_of(j, lane)];
#pragma unroll
    for (int ni = 0; ni < 2; ++ni)
#pragma unroll
      for (int j = 0; j < 16; ++j)
        b[ni][j] = sW[cur][(wn * 32 + ni * 16 + col) * 32 + (lane & 16) + j];

#pragma unroll
    for (int mi = 0; mi < 2; ++mi)
#pragma unroll
      for (int ni = 0; ni < 2; ++ni)
        acc[mi][ni] = __builtin_amdgcn_wmma_f32_16x16x32_bf16(
            false, a[mi], false, b[ni], (short)0, acc[mi][ni], false, false);

    if (it + 1 < KSTEPS) {  // stash staged regs into the other buffer
      uint4* xd = (uint4*)(sX[nxt] + xrow * 32 + xh * 16);
      xd[0] = xr0; xd[1] = xr1;
      *(uint4*)(sW[nxt] + wrow * 32 + wq * 8) = wr0;
    }
    __syncthreads();
  }

#pragma unroll
  for (int mi = 0; mi < 2; ++mi)
#pragma unroll
    for (int ni = 0; ni < 2; ++ni)
#pragma unroll
      for (int r = 0; r < 8; ++r) {
        int m = bm + wm * 32 + mi * 16 + r + hi * 8;
        int n = bn + wn * 32 + ni * 16 + col;
        float v = (acc[mi][ni][r] + bi[ni]) * scale;
        if constexpr (OUTBF)
          ((bf16*)outp)[(size_t)m * NN + n] = (bf16)v;
        else
          ((float*)outp)[(size_t)m * NN + n] = v;
      }
}

// ---------------------------------------------------------------------------
// V pre-transpose: V[b, s, h, d] -> Vt[(b*16+h)*32 + chunk][d][s_in_chunk]
// 64x64 bf16 tiles so the flash PV B-fragment gather becomes contiguous.
// One block per (b, h, chunk). LDS tile row stride 72 keeps b128 stores
// 16B-aligned and spreads the column reads across banks.
// ---------------------------------------------------------------------------
__global__ void __launch_bounds__(256) v_transpose_kernel(
    const unsigned short* __restrict__ V, unsigned short* __restrict__ Vt) {
  __shared__ unsigned short tile[64 * 72];
  const int blk = blockIdx.x;  // ((b*16 + h)*32 + c)
  const int c = blk & 31;
  const int h = (blk >> 5) & 15;
  const int b = blk >> 9;
  const int tid = threadIdx.x;
  {
    int row = tid >> 2, q = tid & 3;  // row = s within chunk
    const uint4* src = (const uint4*)(V + ((size_t)b * Sc + c * 64 + row) * 1024 +
                                      h * 64 + q * 16);
    uint4* dst = (uint4*)(tile + row * 72 + q * 16);
    dst[0] = src[0];
    dst[1] = src[1];
  }
  __syncthreads();
  {
    int d = tid >> 2, sq = tid & 3;
    unsigned w[8];
#pragma unroll
    for (int i = 0; i < 8; ++i) {
      unsigned lo = tile[(sq * 16 + 2 * i + 0) * 72 + d];
      unsigned hh = tile[(sq * 16 + 2 * i + 1) * 72 + d];
      w[i] = lo | (hh << 16);
    }
    uint4* dst = (uint4*)(Vt + (((size_t)blk * 64 + d) * 64) + sq * 16);
    dst[0] = make_uint4(w[0], w[1], w[2], w[3]);
    dst[1] = make_uint4(w[4], w[5], w[6], w[7]);
  }
}

// ---------------------------------------------------------------------------
// Flash attention: one block = (b, h, 128 Q rows). 8 waves x 16 rows.
// Streams S=2048 in 64-column chunks (double-buffered K/V in LDS);
// online softmax; PV accumulated in f32.
// K is [B, S, H, HD] (projection output); V comes pre-transposed as
// [b,h,chunk][d][s] tiles so both B-fragment gathers are contiguous.
// Mask quirk of reference: column index into mask is (s % T).
// ---------------------------------------------------------------------------
__global__ void __launch_bounds__(256) flash_attn_kernel(
    const bf16* __restrict__ Qm, const bf16* __restrict__ Km,
    const bf16* __restrict__ Vt, const float* __restrict__ mask,
    bf16* __restrict__ ctx) {
  __shared__ bf16 sQ[128 * 64];        // 16 KB
  __shared__ bf16 sK[2][64 * 64];      // 16 KB
  __shared__ bf16 sVt[2][64 * 64];     // 16 KB ([d][s] orientation)
  __shared__ bf16 sP[8 * 16 * 64];     // 16 KB (per-wave P staging)

  const int blk = blockIdx.x;
  const int qblk = blk & 7;            // T/128 = 8
  const int h = (blk >> 3) & 15;
  const int b = blk >> 7;
  const int tid = threadIdx.x;
  const int lane = tid & 31;
  const int wid = tid >> 5;
  const int mrow = lane & 15;
  const int hi = (lane >> 4) & 1;
  const int col = lane & 15;

  // Stage 128x64 Q block
  {
    int row = tid >> 1, hf = tid & 1;
    int t = qblk * 128 + row;
    const uint4* src =
        (const uint4*)(Qm + ((size_t)(b * Tc + t)) * Ec + h * 64 + hf * 32);
    uint4* dst = (uint4*)(sQ + row * 64 + hf * 32);
    dst[0] = src[0]; dst[1] = src[1]; dst[2] = src[2]; dst[3] = src[3];
  }

  // K staging: row = s (global layout); V staging: row = d (transposed tiles)
  const int krow = tid >> 2, kq = tid & 3;  // 4 thr/row, 16 bf16 (2x b128)
  const bf16* kbase = Km + ((size_t)b * Sc + krow) * Ec + h * 64 + kq * 16;
  const bf16* vtbase =
      Vt + ((size_t)(b * Hc + h) * 32) * 64 * 64 + krow * 64 + kq * 16;
  uint4 kr0, kr1, vr0, vr1;

  // preload chunk 0
  kr0 = ((const uint4*)kbase)[0]; kr1 = ((const uint4*)kbase)[1];
  vr0 = ((const uint4*)vtbase)[0]; vr1 = ((const uint4*)vtbase)[1];
  {
    uint4* kd = (uint4*)(sK[0] + krow * 64 + kq * 16);
    uint4* vd = (uint4*)(sVt[0] + krow * 64 + kq * 16);
    kd[0] = kr0; kd[1] = kr1;
    vd[0] = vr0; vd[1] = vr1;
  }
  __syncthreads();

  // Per-wave Q fragments: rows wid*16..wid*16+15, HD split in two k=32 steps
  v16bf aq[2];
#pragma unroll
  for (int ks = 0; ks < 2; ++ks)
#pragma unroll
    for (int j = 0; j < 16; ++j)
      aq[ks][j] = sQ[(wid * 16 + mrow) * 64 + ks * 32 + a_k_of(j, lane)];

  // this wave's mask rows start here (row tq = base + r + hi*8)
  const float* mbase =
      mask + ((size_t)b * Tc + qblk * 128 + wid * 16 + hi * 8) * Tc;

  float rmax[8], rsum[8];
#pragma unroll
  for (int r = 0; r < 8; ++r) { rmax[r] = -1e30f; rsum[r] = 0.0f; }
  v8f o[4] = {};

  constexpr int NCHUNK = Sc / 64;  // 32
  for (int it = 0; it < NCHUNK; ++it) {
    const int cur = it & 1, nxt = cur ^ 1;
    const int sc = it * 64;

    if (it + 1 < NCHUNK) {  // issue next chunk's global loads early
      size_t koff = (size_t)(it + 1) * 64 * Ec;
      size_t voff = (size_t)(it + 1) * 64 * 64;
      kr0 = ((const uint4*)(kbase + koff))[0];
      kr1 = ((const uint4*)(kbase + koff))[1];
      vr0 = ((const uint4*)(vtbase + voff))[0];
      vr1 = ((const uint4*)(vtbase + voff))[1];
      if (it + 2 < NCHUNK) {  // L2 prefetch two chunks ahead
        __builtin_prefetch(kbase + koff + (size_t)64 * Ec, 0, 1);
        __builtin_prefetch(vtbase + voff + (size_t)64 * 64, 0, 1);
      }
    }

    // batch all mask loads for this chunk up front (one clause, overlapped)
    float mld[8][4];
#pragma unroll
    for (int r = 0; r < 8; ++r) {
      const float* mp = mbase + (size_t)r * Tc;
#pragma unroll
      for (int nt = 0; nt < 4; ++nt) {
        int s = sc + nt * 16 + col;
        mld[r][nt] = mp[s & (Tc - 1)];
      }
    }

    // logits: 16 x 64 per wave, as 4 n-tiles of 16x16; B(k=d, n=s) = K[s,d]
    v8f lt[4];
#pragma unroll
    for (int nt = 0; nt < 4; ++nt) {
      v8f c = {};
#pragma unroll
      for (int ks = 0; ks < 2; ++ks) {
        v16bf bk;
#pragma unroll
        for (int j = 0; j < 16; ++j)
          bk[j] = sK[cur][(nt * 16 + col) * 64 + ks * 32 + (lane & 16) + j];
        c = __builtin_amdgcn_wmma_f32_16x16x32_bf16(false, aq[ks], false, bk,
                                                    (short)0, c, false, false);
      }
      lt[nt] = c;
    }

    // mask add + online softmax (row m lives across 16 lanes of a half-wave)
#pragma unroll
    for (int r = 0; r < 8; ++r) {
      int m = r + hi * 8;
      float mx = -1e30f;
#pragma unroll
      for (int nt = 0; nt < 4; ++nt) {
        lt[nt][r] += mld[r][nt];
        mx = fmaxf(mx, lt[nt][r]);
      }
#pragma unroll
      for (int off = 1; off < 16; off <<= 1)
        mx = fmaxf(mx, __shfl_xor(mx, off, 32));
      float nm = fmaxf(rmax[r], mx);
      float corr = __expf(rmax[r] - nm);
      rmax[r] = nm;
      float ps = 0.0f;
#pragma unroll
      for (int nt = 0; nt < 4; ++nt) {
        float p = __expf(lt[nt][r] - nm);
        ps += p;
        sP[(wid * 16 + m) * 64 + nt * 16 + col] = (bf16)p;
      }
#pragma unroll
      for (int off = 1; off < 16; off <<= 1)
        ps += __shfl_xor(ps, off, 32);
      rsum[r] = rsum[r] * corr + ps;
#pragma unroll
      for (int dt = 0; dt < 4; ++dt) o[dt][r] *= corr;
    }

    // P (accumulator layout) -> A layout via per-wave LDS round trip
    v16bf ap[2];
#pragma unroll
    for (int ks = 0; ks < 2; ++ks)
#pragma unroll
      for (int j = 0; j < 16; ++j)
        ap[ks][j] = sP[(wid * 16 + mrow) * 64 + ks * 32 + a_k_of(j, lane)];

    // O += P @ V ; B(k=s, n=d) = Vt[d, s] -> contiguous gather in j
#pragma unroll
    for (int dt = 0; dt < 4; ++dt)
#pragma unroll
      for (int ks = 0; ks < 2; ++ks) {
        v16bf bv;
#pragma unroll
        for (int j = 0; j < 16; ++j)
          bv[j] = sVt[cur][(dt * 16 + col) * 64 + ks * 32 + (lane & 16) + j];
        o[dt] = __builtin_amdgcn_wmma_f32_16x16x32_bf16(false, ap[ks], false, bv,
                                                        (short)0, o[dt], false, false);
      }

    if (it + 1 < NCHUNK) {  // stash staged regs into the other buffer
      uint4* kd = (uint4*)(sK[nxt] + krow * 64 + kq * 16);
      uint4* vd = (uint4*)(sVt[nxt] + krow * 64 + kq * 16);
      kd[0] = kr0; kd[1] = kr1;
      vd[0] = vr0; vd[1] = vr1;
    }
    __syncthreads();
  }

  // normalize and write context as [B, T, H*HD] bf16
#pragma unroll
  for (int r = 0; r < 8; ++r) {
    int m = r + hi * 8;
    int tq = qblk * 128 + wid * 16 + m;
    float inv = 1.0f / rsum[r];
#pragma unroll
    for (int dt = 0; dt < 4; ++dt)
      ctx[((size_t)(b * Tc + tq)) * Ec + h * 64 + dt * 16 + col] =
          (bf16)(o[dt][r] * inv);
  }
}

// ---------------------------------------------------------------------------
// Launcher
// ---------------------------------------------------------------------------
extern "C" void kernel_launch(void* const* d_in, const int* in_sizes, int n_in,
                              void* d_out, int out_size, void* d_ws,
                              size_t ws_size, hipStream_t stream) {
  (void)in_sizes; (void)n_in; (void)out_size; (void)ws_size;
  const float* x    = (const float*)d_in[0];
  const float* mask = (const float*)d_in[1];
  const float* Wq   = (const float*)d_in[2];
  const float* bq   = (const float*)d_in[3];
  const float* Wk   = (const float*)d_in[4];
  const float* bk   = (const float*)d_in[5];
  const float* Wv   = (const float*)d_in[6];
  const float* bv   = (const float*)d_in[7];
  const float* Wo   = (const float*)d_in[8];
  const float* bo   = (const float*)d_in[9];
  float* out = (float*)d_out;

  char* ws = (char*)d_ws;
  const size_t MB = 1u << 20;
  bf16* xb  = (bf16*)(ws + 0 * MB);   // x    bf16: 8 MB
  bf16* wqb = (bf16*)(ws + 8 * MB);   // Wq   bf16: 2 MB
  bf16* wkb = (bf16*)(ws + 10 * MB);  // Wk   bf16: 4 MB
  bf16* wvb = (bf16*)(ws + 14 * MB);  // Wv   bf16: 4 MB
  bf16* wob = (bf16*)(ws + 18 * MB);  // Wo   bf16: 2 MB
  bf16* qb  = (bf16*)(ws + 20 * MB);  // Q    bf16: 8 MB
  bf16* kb  = (bf16*)(ws + 28 * MB);  // K    bf16: 16 MB
  bf16* vvb = (bf16*)(ws + 44 * MB);  // V    bf16: 16 MB
  bf16* cb  = (bf16*)(ws + 60 * MB);  // ctx  bf16: 8 MB
  bf16* vtb = (bf16*)(ws + 68 * MB);  // V^T  bf16: 16 MB

  const int M = Bc * Tc;  // 4096 rows for all projections

  auto cvt = [&](const float* s, bf16* d, int n) {
    f32_to_bf16_kernel<<<dim3((n / 4 + 255) / 256), 256, 0, stream>>>(s, d, n);
  };
  cvt(x, xb, M * Ec);
  cvt(Wq, wqb, Ec * Ec);
  cvt(Wk, wkb, 2 * Ec * Ec);
  cvt(Wv, wvb, 2 * Ec * Ec);
  cvt(Wo, wob, Ec * Ec);

  // Q = (x Wq^T + bq) * SCALE   -> bf16 [M, E]
  gemm_bf16_kernel<Ec, true><<<dim3(Ec / 64, M / 128), 256, 0, stream>>>(
      xb, wqb, bq, qb, SCALEc);
  // K = x Wk^T + bk             -> bf16 [M, 2E] == [B, S, H, HD]
  gemm_bf16_kernel<2 * Ec, true><<<dim3(2 * Ec / 64, M / 128), 256, 0, stream>>>(
      xb, wkb, bk, kb, 1.0f);
  // V = x Wv^T + bv
  gemm_bf16_kernel<2 * Ec, true><<<dim3(2 * Ec / 64, M / 128), 256, 0, stream>>>(
      xb, wvb, bv, vvb, 1.0f);

  // V -> per-(b,h,chunk) transposed 64x64 tiles
  v_transpose_kernel<<<dim3(Bc * Hc * (Sc / 64)), 256, 0, stream>>>(
      (const unsigned short*)vvb, (unsigned short*)vtb);

  // fused attention: B*H*(T/128) = 512 blocks
  flash_attn_kernel<<<dim3(Bc * Hc * (Tc / 128)), 256, 0, stream>>>(
      qb, kb, vtb, mask, cb);

  // out = ctx Wo^T + bo  -> f32 [B, T, E]
  gemm_bf16_kernel<Ec, false><<<dim3(Ec / 64, M / 128), 256, 0, stream>>>(
      cb, wob, bo, out, 1.0f);
}
